// brGCN2_3_88467736363031
// MI455X (gfx1250) — compile-verified, matrix operand order checked by reading, not run on hardware
//
#include <hip/hip_runtime.h>

#define DIN  128
#define DHID 128
#define NCLS 64

typedef __attribute__((ext_vector_type(16))) __bf16 v16bf;
typedef __attribute__((ext_vector_type(8)))  float  v8f;

union Frag16 { v16bf v; uint4 q[2]; };

__device__ __forceinline__ unsigned short f2bf(float f) {
  unsigned u = __float_as_uint(f);
  unsigned r = 0x7FFFu + ((u >> 16) & 1u);   // round-to-nearest-even
  return (unsigned short)((u + r) >> 16);
}

static inline int cdiv(long long a, long long b) { return (int)((a + b - 1) / b); }

// ---------------- CSR build: histogram -> exclusive scan -> fill ----------------
__global__ void k_zero_i(int* p, int n) {
  int i = blockIdx.x * blockDim.x + threadIdx.x;
  if (i < n) p[i] = 0;
}

__global__ void k_count(const int* __restrict__ dst, int* __restrict__ cnt, int ne) {
  int i = blockIdx.x * blockDim.x + threadIdx.x;
  if (i < ne) atomicAdd(&cnt[dst[i]], 1);
}

// per-block exclusive scan (256 elems) + block sums
__global__ void k_scan1(const int* __restrict__ in, int* __restrict__ out,
                        int* __restrict__ bsum, int n) {
  __shared__ int tmp[256];
  int t = threadIdx.x;
  int i = blockIdx.x * 256 + t;
  int v = (i < n) ? in[i] : 0;
  tmp[t] = v;
  __syncthreads();
#pragma unroll
  for (int o = 1; o < 256; o <<= 1) {
    int a = (t >= o) ? tmp[t - o] : 0;
    __syncthreads();
    tmp[t] += a;
    __syncthreads();
  }
  if (i < n) out[i] = tmp[t] - v;              // exclusive
  if (t == 255) bsum[blockIdx.x] = tmp[255];   // inclusive block total
}

// single-block exclusive scan of block sums (nb <= 512)
__global__ void k_scan2(int* bsum, int nb) {
  __shared__ int tmp[512];
  int t = threadIdx.x;
  int v = (t < nb) ? bsum[t] : 0;
  tmp[t] = v;
  __syncthreads();
#pragma unroll
  for (int o = 1; o < 512; o <<= 1) {
    int a = (t >= o) ? tmp[t - o] : 0;
    __syncthreads();
    tmp[t] += a;
    __syncthreads();
  }
  if (t < nb) bsum[t] = tmp[t] - v;            // exclusive offsets
}

__global__ void k_scan3(int* __restrict__ rowptr, const int* __restrict__ bsum, int n) {
  int i = blockIdx.x * blockDim.x + threadIdx.x;
  if (i < n) rowptr[i] += bsum[i >> 8];
}

__global__ void k_copy_i(const int* __restrict__ a, int* __restrict__ b, int n) {
  int i = blockIdx.x * blockDim.x + threadIdx.x;
  if (i < n) b[i] = a[i];
}

__global__ void k_fill(const int* __restrict__ src, const int* __restrict__ dst,
                       int* __restrict__ pos, int* __restrict__ col, int ne) {
  int e = blockIdx.x * blockDim.x + threadIdx.x;
  if (e < ne) {
    int slot = atomicAdd(&pos[dst[e]], 1);
    col[slot] = src[e];
  }
}

__global__ void k_dinv(const int* __restrict__ cnt, float* __restrict__ dinv, int n) {
  int i = blockIdx.x * blockDim.x + threadIdx.x;
  if (i < n) dinv[i] = rsqrtf((float)cnt[i] + 1.0f);  // +1 self loop
}

// ---------------- bf16 conversion / weight packing ----------------
__global__ void k_f2bf(const float* __restrict__ in, unsigned short* __restrict__ out, int n) {
  int i = blockIdx.x * blockDim.x + threadIdx.x;
  if (i < n) out[i] = f2bf(in[i]);
}

// Pack row-major W[K x N] (f32) into wave32 WMMA B-fragment order (bf16):
// idx = (((ct*ktiles + kt)*32 + lane)*16 + j)
// lane: col = ct*16 + (lane&15); koff = (lane>>4)*8
// j:    k = kt*32 + koff + j + (j>=8 ? 8 : 0)   (ISA 7.12.2 16-bit layout)
__global__ void k_pack_w(const float* __restrict__ W, unsigned short* __restrict__ Wp,
                         int K, int N, int total) {
  int idx = blockIdx.x * blockDim.x + threadIdx.x;
  if (idx >= total) return;
  int j    = idx & 15;
  int lane = (idx >> 4) & 31;
  int blk  = idx >> 9;            // ct*ktiles + kt
  int ktiles = K >> 5;
  int kt = blk % ktiles;
  int ct = blk / ktiles;
  int koff = (lane >> 4) << 3;
  int k = (kt << 5) + koff + j + ((j >= 8) ? 8 : 0);
  int c = (ct << 4) + (lane & 15);
  Wp[idx] = f2bf(W[(size_t)k * N + c]);
}

// ---------------- WMMA GEMM ----------------
// Each wave owns a distinct 16-row tile and computes ALL N columns:
// A fragment loaded once per k-chunk, reused across N/16 WMMAs; packed B is tiny
// (<=32KB) and stays resident in WGP$.
template <int K, int N>
__global__ __launch_bounds__(256)
void k_gemm(const unsigned short* __restrict__ A, const unsigned short* __restrict__ Bp,
            float* __restrict__ C, int mtiles) {
  const int wave = threadIdx.x >> 5;
  const int lane = threadIdx.x & 31;
  const int tile = blockIdx.x * (blockDim.x >> 5) + wave;
  if (tile >= mtiles) return;
  const int row0 = tile << 4;
  constexpr int KT = K / 32;
  constexpr int CT = N / 16;
  const int r    = lane & 15;
  const int koff = (lane >> 4) << 3;
  const unsigned short* arow  = A + (size_t)(row0 + r) * K + koff;
  const unsigned short* bbase = Bp + ((size_t)lane << 4);

  v8f acc[CT] = {};
#pragma unroll
  for (int kt = 0; kt < KT; ++kt) {
    Frag16 a;
    const uint4* pa = (const uint4*)(arow + (kt << 5));
    a.q[0] = pa[0];                 // K = koff .. koff+7
    a.q[1] = pa[2];                 // K = koff+16 .. koff+23
#pragma unroll
    for (int ct = 0; ct < CT; ++ct) {
      Frag16 b;
      const uint4* pb = (const uint4*)(bbase + (((size_t)(ct * KT + kt)) << 9));
      b.q[0] = pb[0];
      b.q[1] = pb[1];
      acc[ct] = __builtin_amdgcn_wmma_f32_16x16x32_bf16(
          false, a.v, false, b.v, (short)0, acc[ct], false, false);
    }
  }
  // C/D layout: VGPR v -> row (lane<16 ? v : v+8), col = lane&15
  const int orow = row0 + ((lane >> 4) << 3);
#pragma unroll
  for (int ct = 0; ct < CT; ++ct) {
    float* cp = C + (size_t)orow * N + (ct << 4) + r;
#pragma unroll
    for (int v = 0; v < 8; ++v) cp[(size_t)v * N] = acc[ct][v];
  }
}

// ---------------- gather aggregation (CSR), one wave per destination node ----------
// layer 1: out_bf16 = bf16(relu(b1 + dinv[i]^2*h[i] + sum_j dinv[i]*dinv[src_j]*h[src_j]))
__global__ void k_agg1(const float* __restrict__ h, const int* __restrict__ rowptr,
                       const int* __restrict__ cnt, const int* __restrict__ col,
                       const float* __restrict__ dinv, const float* __restrict__ bias,
                       unsigned short* __restrict__ out, int n) {
  int node = (blockIdx.x * blockDim.x + threadIdx.x) >> 5;
  int lane = threadIdx.x & 31;
  if (node >= n) return;
  float di = dinv[node];
  float4 hv = *((const float4*)(h + (size_t)node * DHID) + lane);
  float4 bv = *((const float4*)bias + lane);
  float w0 = di * di;
  float4 acc;
  acc.x = bv.x + w0 * hv.x; acc.y = bv.y + w0 * hv.y;
  acc.z = bv.z + w0 * hv.z; acc.w = bv.w + w0 * hv.w;
  int beg = rowptr[node], m = cnt[node];
  for (int j0 = 0; j0 < m; j0 += 32) {
    int rem = m - j0;
    int c  = (lane < rem) ? col[beg + j0 + lane] : 0;      // coalesced batch load
    float dw = (lane < rem) ? dinv[c] : 0.0f;
    int kmax = rem < 32 ? rem : 32;
    for (int k = 0; k < kmax; ++k) {
      int s   = __shfl(c, k, 32);
      float w = di * __shfl(dw, k, 32);
      float4 v = *((const float4*)(h + (size_t)s * DHID) + lane);  // 512B coalesced, L2-resident
      acc.x += w * v.x; acc.y += w * v.y; acc.z += w * v.z; acc.w += w * v.w;
    }
  }
  ushort4 o;
  o.x = f2bf(fmaxf(acc.x, 0.f)); o.y = f2bf(fmaxf(acc.y, 0.f));
  o.z = f2bf(fmaxf(acc.z, 0.f)); o.w = f2bf(fmaxf(acc.w, 0.f));
  *((ushort4*)(out + (size_t)node * DHID) + lane) = o;
}

// layer 2: aggregation + bias + fused wave-level log-softmax (64 classes, 2/lane)
__global__ void k_agg2(const float* __restrict__ h, const int* __restrict__ rowptr,
                       const int* __restrict__ cnt, const int* __restrict__ col,
                       const float* __restrict__ dinv, const float* __restrict__ bias,
                       float* __restrict__ out, int n) {
  int node = (blockIdx.x * blockDim.x + threadIdx.x) >> 5;
  int lane = threadIdx.x & 31;
  if (node >= n) return;
  float di = dinv[node];
  float2 hv = *((const float2*)(h + (size_t)node * NCLS) + lane);
  float2 bv = *((const float2*)bias + lane);
  float w0 = di * di;
  float2 acc;
  acc.x = bv.x + w0 * hv.x; acc.y = bv.y + w0 * hv.y;
  int beg = rowptr[node], m = cnt[node];
  for (int j0 = 0; j0 < m; j0 += 32) {
    int rem = m - j0;
    int c  = (lane < rem) ? col[beg + j0 + lane] : 0;
    float dw = (lane < rem) ? dinv[c] : 0.0f;
    int kmax = rem < 32 ? rem : 32;
    for (int k = 0; k < kmax; ++k) {
      int s   = __shfl(c, k, 32);
      float w = di * __shfl(dw, k, 32);
      float2 v = *((const float2*)(h + (size_t)s * NCLS) + lane);
      acc.x += w * v.x; acc.y += w * v.y;
    }
  }
  float mx = fmaxf(acc.x, acc.y);
#pragma unroll
  for (int o = 16; o; o >>= 1) mx = fmaxf(mx, __shfl_xor(mx, o, 32));
  float s = __expf(acc.x - mx) + __expf(acc.y - mx);
#pragma unroll
  for (int o = 16; o; o >>= 1) s += __shfl_xor(s, o, 32);
  float l = mx + __logf(s);
  float2 rr; rr.x = acc.x - l; rr.y = acc.y - l;
  *((float2*)(out + (size_t)node * NCLS) + lane) = rr;
}

// ---------------- launch ----------------
extern "C" void kernel_launch(void* const* d_in, const int* in_sizes, int n_in,
                              void* d_out, int out_size, void* d_ws, size_t ws_size,
                              hipStream_t stream) {
  const float* x  = (const float*)d_in[0];
  const int*   ei = (const int*)d_in[1];
  const float* W1 = (const float*)d_in[2];
  const float* b1 = (const float*)d_in[3];
  const float* W2 = (const float*)d_in[4];
  const float* b2 = (const float*)d_in[5];

  const int n  = in_sizes[0] / DIN;   // 100000
  const int ne = in_sizes[1] / 2;     // 1600000
  const int* src = ei;
  const int* dst = ei + ne;

  char* ws = (char*)d_ws;
  size_t off = 0;
  auto take = [&](size_t bytes) -> char* {
    char* p = ws + off;
    off += (bytes + 255) & ~(size_t)255;
    return p;
  };
  int*            cnt    = (int*)take((size_t)n * 4);
  int*            rowptr = (int*)take((size_t)n * 4);
  int*            pos    = (int*)take((size_t)n * 4);
  int*            bsum   = (int*)take(512 * 4);
  int*            col    = (int*)take((size_t)ne * 4);
  float*          dinv   = (float*)take((size_t)n * 4);
  unsigned short* xbf    = (unsigned short*)take((size_t)n * DIN * 2);  // reused: relu(h1) bf16
  float*          h      = (float*)take((size_t)n * DIN * 4);           // GEMM out (both layers)
  unsigned short* w1p    = (unsigned short*)take((size_t)(DHID / 16) * (DIN / 32) * 512 * 2);
  unsigned short* w2p    = (unsigned short*)take((size_t)(NCLS / 16) * (DHID / 32) * 512 * 2);
  float* out = (float*)d_out;

  const int T = 256;
  const int nb = cdiv(n, 256);   // scan blocks (391 <= 512)

  // ---- CSR by destination + normalization ----
  k_zero_i<<<cdiv(n, T), T, 0, stream>>>(cnt, n);
  k_count<<<cdiv(ne, T), T, 0, stream>>>(dst, cnt, ne);
  k_scan1<<<nb, 256, 0, stream>>>(cnt, rowptr, bsum, n);
  k_scan2<<<1, 512, 0, stream>>>(bsum, nb);
  k_scan3<<<cdiv(n, T), T, 0, stream>>>(rowptr, bsum, n);
  k_copy_i<<<cdiv(n, T), T, 0, stream>>>(rowptr, pos, n);
  k_fill<<<cdiv(ne, T), T, 0, stream>>>(src, dst, pos, col, ne);
  k_dinv<<<cdiv(n, T), T, 0, stream>>>(cnt, dinv, n);

  // ---- precision conversion + WMMA fragment packing ----
  k_f2bf<<<cdiv((long long)n * DIN, T), T, 0, stream>>>(x, xbf, n * DIN);
  {
    int t1 = (DHID / 16) * (DIN / 32) * 512;
    int t2 = (NCLS / 16) * (DHID / 32) * 512;
    k_pack_w<<<cdiv(t1, T), T, 0, stream>>>(W1, w1p, DIN, DHID, t1);
    k_pack_w<<<cdiv(t2, T), T, 0, stream>>>(W2, w2p, DHID, NCLS, t2);
  }

  const int mtiles = cdiv(n, 16);  // 6250

  // ---- layer 1: GEMM + gather-aggregate + fused relu/bf16 ----
  k_gemm<DIN, DHID><<<cdiv(mtiles, 8), 256, 0, stream>>>(xbf, w1p, h, mtiles);
  k_agg1<<<cdiv((long long)n * 32, T), T, 0, stream>>>(h, rowptr, cnt, col, dinv, b1, xbf, n);

  // ---- layer 2: GEMM + gather-aggregate + fused log-softmax ----
  k_gemm<DHID, NCLS><<<cdiv(mtiles, 8), 256, 0, stream>>>(xbf, w2p, h, mtiles);
  k_agg2<<<cdiv((long long)n * 32, T), T, 0, stream>>>(h, rowptr, cnt, col, dinv, b2, out, n);
}
